// Attention_36172214567800
// MI455X (gfx1250) — compile-verified, matrix operand order checked by reading, not run on hardware
//
#include <hip/hip_runtime.h>
#include <hip/hip_bf16.h>

// ---------------------------------------------------------------------------
// Causal MHA for MI455X (gfx1250, wave32, WMMA f32_16x16x32_f16).
// B=4, N=2048, D=1024, H=16, DH=64.
// Pipeline: f32->f16 convert -> QKV GEMM (LDS-staged weights via async-LDS
// loads, head-scatter, V transposed) -> flash attention (online softmax, LDS
// P-restage) -> output projection GEMM (f32 out).
// ---------------------------------------------------------------------------

typedef __attribute__((ext_vector_type(16))) _Float16 v16h;
typedef __attribute__((ext_vector_type(8)))  _Float16 v8h;
typedef __attribute__((ext_vector_type(8)))  float    v8f;

#define WMMA16(a, b, c) \
  __builtin_amdgcn_wmma_f32_16x16x32_f16(false, (a), false, (b), (short)0, (c), false, false)

#define BB 4
#define NN 2048
#define DD 1024
#define HH 16
#define DH 64
#define MASKV (-3.0e38f)

// 16-byte per-lane global->LDS async copy (ASYNCcnt-tracked, ISA §15.18.3
// op 98). Inline asm sidesteps toolchain-specific builtin signatures.
// Flat LDS pointer low 32 bits == LDS byte offset (ISA §10.2 aperture table);
// flat global pointer value == global address.
__device__ __forceinline__ void cp16_g2l(const _Float16* g, _Float16* l) {
  const unsigned lds32 = (unsigned)(size_t)(const void*)l;
  const unsigned long long ga = (unsigned long long)(size_t)(const void*)g;
  asm volatile("global_load_async_to_lds_b128 %0, %1, off"
               :: "v"(lds32), "v"(ga) : "memory");
}
__device__ __forceinline__ void cp_g2l_wait() {
#if __has_builtin(__builtin_amdgcn_s_wait_asynccnt)
  __builtin_amdgcn_s_wait_asynccnt(0);
#else
  asm volatile("s_wait_asynccnt 0x0" ::: "memory");
#endif
}

// Load a 16x32 f16 fragment from a row-major matrix (leading dim `ld` in
// elements). Per CDNA5 ISA 7.12.2: lane l holds row (l&15); K offsets
// kb..kb+7 in halves 0..7 and kb+16..kb+23 in halves 8..15, kb = 8*(l>>4).
// Same addressing serves the B-fragment of A*B^T (column n = row n of B^T).
// Works for both global and LDS base pointers (addrspace inferred inline).
__device__ __forceinline__ v16h load_frag(const _Float16* __restrict__ base,
                                          int ld, int lane) {
  const int m  = lane & 15;
  const int kb = (lane >> 4) << 3;
  const _Float16* p = base + (size_t)m * ld + kb;
  v8h lo = *(const v8h*)(p);
  v8h hi = *(const v8h*)(p + 16);
  v16h r;
#pragma unroll
  for (int i = 0; i < 8; ++i) { r[i] = lo[i]; r[i + 8] = hi[i]; }
  return r;
}

// Row reductions across the 16 lanes that hold one C-row (wave32 halves).
__device__ __forceinline__ float rmax16(float v) {
#pragma unroll
  for (int m = 1; m < 16; m <<= 1) v = fmaxf(v, __shfl_xor(v, m, 32));
  return v;
}
__device__ __forceinline__ float rsum16(float v) {
#pragma unroll
  for (int m = 1; m < 16; m <<= 1) v += __shfl_xor(v, m, 32);
  return v;
}

// ---------------------------------------------------------------------------
__global__ void cvt_f32_f16(const float* __restrict__ s,
                            _Float16* __restrict__ d, int n) {
  int i = blockIdx.x * blockDim.x + threadIdx.x;
  if (i < n) d[i] = (_Float16)s[i];
}

// ---------------------------------------------------------------------------
// QKV projection: C[m,e] = sum_d X[m,d] * Wqkv[e,d].
// 4 waves/block; each wave owns a 16-row m-tile; the block shares a 64x32
// weight slab staged in LDS by async global->LDS loads each k-step.
// Scatter epilogue: e<1024 -> Q[b,h,n,dh]; e<2048 -> K; else V^T[b,h,dh,n].
__global__ __launch_bounds__(128) void qkv_gemm(
    const _Float16* __restrict__ xh, const _Float16* __restrict__ wh,
    _Float16* __restrict__ qh, _Float16* __restrict__ kh,
    _Float16* __restrict__ vth) {
  __shared__ __align__(16) _Float16 sW[64 * 32];  // 4 KB W slab
  const int tid  = threadIdx.x;
  const int lane = tid & 31;
  const int wv   = tid >> 5;
  const int e0 = blockIdx.x * 64;
  const int m0 = (blockIdx.y * 4 + wv) * 16;
  const _Float16* abase = xh + (size_t)m0 * DD;

  v8f c[4];
#pragma unroll
  for (int s = 0; s < 4; ++s)
#pragma unroll
    for (int r = 0; r < 8; ++r) c[s][r] = 0.f;

  for (int k = 0; k < DD; k += 32) {
    __syncthreads();  // WAR: previous slab fully consumed
#pragma unroll
    for (int q = tid; q < 256; q += 128) {   // 256 x 16B chunks
      const int row = q >> 2;                // 0..63 (e offset)
      const int off = (q & 3) * 8;           // halves within row
      cp16_g2l(wh + (size_t)(e0 + row) * DD + k + off, sW + row * 32 + off);
    }
    cp_g2l_wait();
    __syncthreads();  // RAW: slab visible to all waves

    __builtin_prefetch(abase + k + 64, 0, 1);  // -> global_prefetch_b8
    v16h a = load_frag(abase + k, DD, lane);
#pragma unroll
    for (int s = 0; s < 4; ++s) {
      v16h b = load_frag(sW + 16 * s * 32, 32, lane);  // ds_load_b128 x2
      c[s] = WMMA16(a, b, c[s]);
    }
  }

  const int hw = lane >> 4, col = lane & 15;
#pragma unroll
  for (int s = 0; s < 4; ++s) {
#pragma unroll
    for (int r = 0; r < 8; ++r) {
      const int e   = e0 + 16 * s + col;
      const int row = m0 + r + 8 * hw;          // global b*N + n
      const int b   = row >> 11;                // /2048
      const int n   = row & (NN - 1);
      const _Float16 hv = (_Float16)c[s][r];
      if (e < DD) {
        const int h = e >> 6, dh = e & 63;
        qh[(((size_t)(b * HH + h) * NN + n) << 6) + dh] = hv;
      } else if (e < 2 * DD) {
        const int e2 = e - DD, h = e2 >> 6, dh = e2 & 63;
        kh[(((size_t)(b * HH + h) * NN + n) << 6) + dh] = hv;
      } else {
        const int e2 = e - 2 * DD, h = e2 >> 6, dh = e2 & 63;
        vth[((size_t)(b * HH + h) * DH + dh) * NN + n] = hv;  // transposed
      }
    }
  }
}

// ---------------------------------------------------------------------------
// Flash attention: one wave per 16-query tile; key tiles of 32 (two 16-wide
// score tiles per step); causal bound on the loop; P restaged via LDS.
__global__ __launch_bounds__(128) void attn_kernel(
    const _Float16* __restrict__ qh, const _Float16* __restrict__ kh,
    const _Float16* __restrict__ vth, _Float16* __restrict__ aoh) {
  __shared__ __align__(16) _Float16 sP[4][16][32];  // per-wave P tile

  const int lane = threadIdx.x & 31;
  const int wv   = threadIdx.x >> 5;
  const int bh   = blockIdx.x;            // b*H + h
  const int b    = bh >> 4, h = bh & 15;
  const int it   = blockIdx.y * 4 + wv;   // query tile index, 0..127
  const int n0   = it * 16;

  const _Float16* qb = qh  + (size_t)bh * NN * DH;
  const _Float16* kb = kh  + (size_t)bh * NN * DH;
  const _Float16* vb = vth + (size_t)bh * DH * NN;

  const v16h aq0 = load_frag(qb + (size_t)n0 * DH +  0, DH, lane);
  const v16h aq1 = load_frag(qb + (size_t)n0 * DH + 32, DH, lane);

  v8f o[4];
  float mrun[8], lrun[8];
#pragma unroll
  for (int r = 0; r < 8; ++r) { mrun[r] = MASKV; lrun[r] = 0.f; }
#pragma unroll
  for (int s = 0; s < 4; ++s)
#pragma unroll
    for (int r = 0; r < 8; ++r) o[s][r] = 0.f;

  const int hw = lane >> 4, col = lane & 15;
  const int npairs = (it >> 1) + 1;       // causal: keys <= n0+15 covered

  for (int p = 0; p < npairs; ++p) {
    const int k0 = p * 32;
    // ---- S = (Q K^T) * 1/sqrt(DH): two 16x16 score tiles ----
    v8f s0, s1;
#pragma unroll
    for (int r = 0; r < 8; ++r) { s0[r] = 0.f; s1[r] = 0.f; }
    {
      v16h bk = load_frag(kb + (size_t)k0 * DH, DH, lane);
      s0 = WMMA16(aq0, bk, s0);
      bk = load_frag(kb + (size_t)k0 * DH + 32, DH, lane);
      s0 = WMMA16(aq1, bk, s0);
      bk = load_frag(kb + (size_t)(k0 + 16) * DH, DH, lane);
      s1 = WMMA16(aq0, bk, s1);
      bk = load_frag(kb + (size_t)(k0 + 16) * DH + 32, DH, lane);
      s1 = WMMA16(aq1, bk, s1);
    }
    // ---- online softmax over the 32 new keys ----
#pragma unroll
    for (int r = 0; r < 8; ++r) {
      const int row = n0 + r + 8 * hw;
      float a0 = s0[r] * 0.125f;
      float a1 = s1[r] * 0.125f;
      if (k0 + col      > row) a0 = MASKV;   // causal mask
      if (k0 + 16 + col > row) a1 = MASKV;
      const float mnew = fmaxf(mrun[r], rmax16(fmaxf(a0, a1)));
      const float corr = __expf(mrun[r] - mnew);
      mrun[r] = mnew;
      const float p0 = __expf(a0 - mnew);
      const float p1 = __expf(a1 - mnew);
      lrun[r] = lrun[r] * corr + rsum16(p0 + p1);
#pragma unroll
      for (int s = 0; s < 4; ++s) o[s][r] *= corr;
      sP[wv][r + 8 * hw][col]      = (_Float16)p0;   // C-layout -> LDS
      sP[wv][r + 8 * hw][col + 16] = (_Float16)p1;
    }
    // ---- O += P V (same-wave LDS ordering keeps RAW safe) ----
    const v16h pf = load_frag(&sP[wv][0][0], 32, lane);  // ds_load_b128 x2
#pragma unroll
    for (int s = 0; s < 4; ++s) {
      v16h bv = load_frag(vb + (size_t)(16 * s) * NN + k0, NN, lane);
      o[s] = WMMA16(pf, bv, o[s]);
    }
  }

  // ---- normalize and write [B*N, D] head-interleaved f16 ----
#pragma unroll
  for (int r = 0; r < 8; ++r) {
    const float inv = 1.0f / lrun[r];
    const size_t row = (size_t)b * NN + n0 + r + 8 * hw;
#pragma unroll
    for (int s = 0; s < 4; ++s)
      aoh[row * DD + h * DH + 16 * s + col] = (_Float16)(o[s][r] * inv);
  }
}

// ---------------------------------------------------------------------------
// Output projection: out[m,e] = sum_d AO[m,d] * Wout[e,d]; f32 result.
// Same 4-wave LDS-staged structure as qkv_gemm.
__global__ __launch_bounds__(128) void out_gemm(
    const _Float16* __restrict__ aoh, const _Float16* __restrict__ wh,
    float* __restrict__ out) {
  __shared__ __align__(16) _Float16 sW[64 * 32];
  const int tid  = threadIdx.x;
  const int lane = tid & 31;
  const int wv   = tid >> 5;
  const int e0 = blockIdx.x * 64;
  const int m0 = (blockIdx.y * 4 + wv) * 16;
  const _Float16* abase = aoh + (size_t)m0 * DD;

  v8f c[4];
#pragma unroll
  for (int s = 0; s < 4; ++s)
#pragma unroll
    for (int r = 0; r < 8; ++r) c[s][r] = 0.f;

  for (int k = 0; k < DD; k += 32) {
    __syncthreads();
#pragma unroll
    for (int q = tid; q < 256; q += 128) {
      const int row = q >> 2;
      const int off = (q & 3) * 8;
      cp16_g2l(wh + (size_t)(e0 + row) * DD + k + off, sW + row * 32 + off);
    }
    cp_g2l_wait();
    __syncthreads();

    __builtin_prefetch(abase + k + 64, 0, 1);
    v16h a = load_frag(abase + k, DD, lane);
#pragma unroll
    for (int s = 0; s < 4; ++s) {
      v16h b = load_frag(sW + 16 * s * 32, 32, lane);
      c[s] = WMMA16(a, b, c[s]);
    }
  }

  const int hw = lane >> 4, col = lane & 15;
#pragma unroll
  for (int s = 0; s < 4; ++s)
#pragma unroll
    for (int r = 0; r < 8; ++r)
      out[(size_t)(m0 + r + 8 * hw) * DD + e0 + 16 * s + col] = c[s][r];
}

// ---------------------------------------------------------------------------
extern "C" void kernel_launch(void* const* d_in, const int* in_sizes, int n_in,
                              void* d_out, int out_size, void* d_ws,
                              size_t ws_size, hipStream_t stream) {
  (void)in_sizes; (void)n_in; (void)out_size; (void)ws_size;
  const float* x    = (const float*)d_in[0];
  // d_in[1] is the causal mask [1,1,N,N]; masking is computed analytically.
  const float* wqkv = (const float*)d_in[2];
  const float* wout = (const float*)d_in[3];
  float* out = (float*)d_out;

  const size_t XE = (size_t)BB * NN * DD;     // 8,388,608 elements
  _Float16* ws    = (_Float16*)d_ws;
  _Float16* xh    = ws;  ws += XE;
  _Float16* wqkvh = ws;  ws += (size_t)3 * DD * DD;
  _Float16* wouth = ws;  ws += (size_t)DD * DD;
  _Float16* qh    = ws;  ws += XE;
  _Float16* kh    = ws;  ws += XE;
  _Float16* vth   = ws;  ws += XE;
  _Float16* aoh   = ws;  ws += XE;

  const int n1 = (int)XE, n2 = 3 * DD * DD, n3 = DD * DD;
  cvt_f32_f16<<<(n1 + 255) / 256, 256, 0, stream>>>(x, xh, n1);
  cvt_f32_f16<<<(n2 + 255) / 256, 256, 0, stream>>>(wqkv, wqkvh, n2);
  cvt_f32_f16<<<(n3 + 255) / 256, 256, 0, stream>>>(wout, wouth, n3);

  dim3 gq(3 * DD / 64, BB * NN / 64);               // 48 x 128, 4 waves/block
  qkv_gemm<<<gq, 128, 0, stream>>>(xh, wqkvh, qh, kh, vth);

  dim3 ga(BB * HH, NN / 64);                        // 64 x 32, 4 waves/block
  attn_kernel<<<ga, 128, 0, stream>>>(qh, kh, vth, aoh);

  dim3 go(DD / 64, BB * NN / 64);                   // 16 x 128, 4 waves/block
  out_gemm<<<go, 128, 0, stream>>>(aoh, wouth, out);
}
